// RNN_52183852646802
// MI455X (gfx1250) — matrix operand order repaired and checked by reading, not compiled, not run on hardware
//
#include <hip/hip_runtime.h>
#include <hip/hip_bf16.h>

#define VOCAB   50257
#define D_EMB   128
#define HIDDEN  256
#define N_CLS   2
#define BATCH_B 256
#define SEQ_T   2048
#define MB      16      // batch rows per workgroup

typedef __attribute__((ext_vector_type(16))) __bf16 v16bf;
typedef __attribute__((ext_vector_type(8)))  __bf16 v8bf;
typedef __attribute__((ext_vector_type(8)))  float  v8f;

static __device__ __forceinline__ v8f wmma_bf16(v16bf a, v16bf b, v8f c) {
    // D = A(16x32 bf16) * B(32x16 bf16) + C(16x16 f32)
    return __builtin_amdgcn_wmma_f32_16x16x32_bf16(
        /*neg_a=*/false, a, /*neg_b=*/false, b,
        /*c_mod=*/(short)0, c, /*reuse_a=*/false, /*reuse_b=*/false);
}

static __device__ __forceinline__ float fast_tanh(float x) {
    float e = __expf(2.0f * x);
    return 1.0f - 2.0f * __builtin_amdgcn_rcpf(e + 1.0f);
}

__global__ void __launch_bounds__(256)
rnn_fused(const int* __restrict__ batch,
          const float* __restrict__ emb,
          const float* __restrict__ wx_w, const float* __restrict__ wx_b,
          const float* __restrict__ wh_w, const float* __restrict__ wh_b,
          const float* __restrict__ clf_w, float* __restrict__ out)
{
    // double-buffered hidden state, bf16, row-major [m][k]
    __shared__ __align__(16) __bf16 hbuf[2][MB * HIDDEN];

    const int tid  = threadIdx.x;
    const int wave = tid >> 5;
    const int lane = tid & 31;
    const int ln   = lane & 15;     // A-row / N-column sub-index
    const int hi   = lane >> 4;     // half-wave select
    const int rowbase = blockIdx.x * MB;

    // h(0) = 0
    for (int i = tid; i < MB * HIDDEN; i += 256) hbuf[0][i] = (__bf16)0.0f;

    // this wave's two output column tiles
    const int n0 = wave * 32 + ln;
    const int n1 = n0 + 16;

    // ---- persistent B fragments (weights) in VGPRs, built once ----
    // B-frag layout (ISA 7.12.2): lane -> N = lane&15 ; element i -> K = 32*kc + 8*hi + (i&7) + 16*(i>>3)
    v16bf Bh[8][2];
    v16bf Bx[4][2];
#pragma unroll
    for (int kc = 0; kc < 8; ++kc) {
        const int kb = kc * 32 + hi * 8;
#pragma unroll
        for (int i = 0; i < 16; ++i) {
            Bh[kc][0][i] = (__bf16)wh_w[(kb + (i & 7) + 16 * (i >> 3)) * HIDDEN + n0];
            Bh[kc][1][i] = (__bf16)wh_w[(kb + (i & 7) + 16 * (i >> 3)) * HIDDEN + n1];
        }
    }
#pragma unroll
    for (int kc = 0; kc < 4; ++kc) {
        const int kb = kc * 32 + hi * 8;
#pragma unroll
        for (int i = 0; i < 16; ++i) {
            Bx[kc][0][i] = (__bf16)wx_w[(kb + (i & 7) + 16 * (i >> 3)) * HIDDEN + n0];
            Bx[kc][1][i] = (__bf16)wx_w[(kb + (i & 7) + 16 * (i >> 3)) * HIDDEN + n1];
        }
    }

    const float bias0 = wx_b[n0] + wh_b[n0];
    const float bias1 = wx_b[n1] + wh_b[n1];
    const float cw00 = clf_w[n0 * 2 + 0], cw01 = clf_w[n0 * 2 + 1];
    const float cw10 = clf_w[n1 * 2 + 0], cw11 = clf_w[n1 * 2 + 1];

    // token stream for this lane's A-matrix row (m = ln)
    const int* tokp = batch + (long)(rowbase + ln) * SEQ_T;

    __syncthreads();

    for (int t = 0; t < SEQ_T; ++t) {
        // =========== phase 1: materialize all 12 A-fragments ===========
        // A-frag (16-bit, 16x32): lane -> M = lane&15 ;
        //   elem i -> k = 32*kc + 8*hi + (i&7) + 16*(i>>3)
        v16bf afrag[12];

        // x chunks (0..3) straight from global gather (f32 -> bf16)
        const int tok = tokp[t];
        const float* xr = emb + (long)tok * D_EMB;
#pragma unroll
        for (int kc = 0; kc < 4; ++kc) {
            const int d0 = kc * 32 + hi * 8;
            const float4 f0 = *(const float4*)(xr + d0);
            const float4 f1 = *(const float4*)(xr + d0 + 4);
            const float4 f2 = *(const float4*)(xr + d0 + 16);
            const float4 f3 = *(const float4*)(xr + d0 + 20);
            v16bf a;
            a[0]  = (__bf16)f0.x; a[1]  = (__bf16)f0.y; a[2]  = (__bf16)f0.z; a[3]  = (__bf16)f0.w;
            a[4]  = (__bf16)f1.x; a[5]  = (__bf16)f1.y; a[6]  = (__bf16)f1.z; a[7]  = (__bf16)f1.w;
            a[8]  = (__bf16)f2.x; a[9]  = (__bf16)f2.y; a[10] = (__bf16)f2.z; a[11] = (__bf16)f2.w;
            a[12] = (__bf16)f3.x; a[13] = (__bf16)f3.y; a[14] = (__bf16)f3.z; a[15] = (__bf16)f3.w;
            afrag[kc] = a;
        }

        // h chunks (4..11) from LDS (two ds_load_b128 per chunk)
        const __bf16* hb = &hbuf[t & 1][ln * HIDDEN];
#pragma unroll
        for (int kc = 0; kc < 8; ++kc) {
            const int k0 = kc * 32 + hi * 8;
            v8bf lo = *(const v8bf*)(hb + k0);
            v8bf hh = *(const v8bf*)(hb + k0 + 16);
            afrag[4 + kc] = __builtin_shufflevector(lo, hh,
                              0,1,2,3,4,5,6,7,8,9,10,11,12,13,14,15);
        }

        // prefetch next step's embedding rows into caches
        if (t + 1 < SEQ_T) {
            const int tok2 = tokp[t + 1];
            const char* pf = (const char*)(emb + (long)tok2 * D_EMB);
            __builtin_prefetch(pf + hi * 128, 0, 3);          // global_prefetch_b8
            __builtin_prefetch(pf + 256 + hi * 128, 0, 3);
        }

        // =========== phase 2: 24 back-to-back WMMAs ===========
        // C init with fused bias (bias depends only on N -> broadcast over M)
        v8f c0, c1;
#pragma unroll
        for (int r = 0; r < 8; ++r) { c0[r] = bias0; c1[r] = bias1; }

#pragma unroll
        for (int kc = 0; kc < 4; ++kc) {
            c0 = wmma_bf16(afrag[kc], Bx[kc][0], c0);
            c1 = wmma_bf16(afrag[kc], Bx[kc][1], c1);
        }
#pragma unroll
        for (int kc = 0; kc < 8; ++kc) {
            c0 = wmma_bf16(afrag[4 + kc], Bh[kc][0], c0);
            c1 = wmma_bf16(afrag[4 + kc], Bh[kc][1], c1);
        }

        // =========== phase 3: tanh + write next h (classifier on last step) ===========
        // D layout: VGPR r -> M = r + 8*hi ; lane -> N = lane&15 (+tile offset)
        __bf16* ho = hbuf[(t + 1) & 1];
        if (t != SEQ_T - 1) {
#pragma unroll
            for (int r = 0; r < 8; ++r) {
                const int m = r + hi * 8;
                ho[m * HIDDEN + n0] = (__bf16)fast_tanh(c0[r]);
                ho[m * HIDDEN + n1] = (__bf16)fast_tanh(c1[r]);
            }
        } else {
#pragma unroll
            for (int r = 0; r < 8; ++r) {
                const int m = r + hi * 8;
                const float th0 = fast_tanh(c0[r]);
                const float th1 = fast_tanh(c1[r]);
                atomicAdd(&out[(rowbase + m) * 2 + 0], th0 * cw00 + th1 * cw10);
                atomicAdd(&out[(rowbase + m) * 2 + 1], th0 * cw01 + th1 * cw11);
            }
        }
        __syncthreads();
    }
}

__global__ void init_out_kernel(const float* __restrict__ clf_b, float* __restrict__ out) {
    int i = blockIdx.x * 256 + threadIdx.x;
    if (i < BATCH_B * N_CLS) out[i] = clf_b[i & 1];
}

extern "C" void kernel_launch(void* const* d_in, const int* in_sizes, int n_in,
                              void* d_out, int out_size, void* d_ws, size_t ws_size,
                              hipStream_t stream) {
    (void)in_sizes; (void)n_in; (void)out_size; (void)d_ws; (void)ws_size;
    const int*   batch = (const int*)  d_in[0];
    const float* emb   = (const float*)d_in[1];
    const float* wx_w  = (const float*)d_in[2];
    const float* wx_b  = (const float*)d_in[3];
    const float* wh_w  = (const float*)d_in[4];
    const float* wh_b  = (const float*)d_in[5];
    const float* clf_w = (const float*)d_in[6];
    const float* clf_b = (const float*)d_in[7];
    float* out = (float*)d_out;

    init_out_kernel<<<2, 256, 0, stream>>>(clf_b, out);
    rnn_fused<<<BATCH_B / MB, 256, 0, stream>>>(batch, emb, wx_w, wx_b,
                                                wh_w, wh_b, clf_w, out);
}
